// E_MHSA_78323023610067
// MI455X (gfx1250) — compile-verified, matrix operand order checked by reading, not run on hardware
//
#include <hip/hip_runtime.h>
#include <hip/hip_bf16.h>

typedef __attribute__((ext_vector_type(16))) _Float16 v16h;
typedef __attribute__((ext_vector_type(8)))  _Float16 v8h;
typedef __attribute__((ext_vector_type(8)))  float    v8f;
typedef __attribute__((ext_vector_type(4)))  int      v4i;

#define DIM 384
#define HD 32
#define NH 12
#define NB 8
#define NTOK 1024
#define SCALE 0.17677669529663687f  // 32^-0.5

#if defined(__AMDGCN__) && __has_builtin(__builtin_amdgcn_global_load_async_to_lds_b128)
#define USE_ASYNC_LDS 1
typedef __attribute__((address_space(1))) v4i gv4i;   // global int32x4
typedef __attribute__((address_space(3))) v4i lv4i;   // LDS int32x4
#else
#define USE_ASYNC_LDS 0
#endif

// ---------------------------------------------------------------------------
// A-operand loader for v_wmma_f32_16x16x32_f16, from row-major f16 (row=M,
// contiguous K). Per ISA 7.12.2: lanes 0-15 hold M=lane, K=0..7 in VGPR0-3 and
// K=16..23 in VGPR4-7; lanes 16-31 hold K=8..15 / K=24..31.
// The f16 B-operand layout is the mirror (lane=N), so the SAME loader reads a
// B tile from transposed-row-major memory (row=N, contiguous K).
// ---------------------------------------------------------------------------
__device__ __forceinline__ v16h load_tile16(const _Float16* base, int stride) {
  int lane = threadIdx.x & 31;
  const _Float16* p = base + (lane & 15) * stride + (lane >> 4) * 8;
  v8h lo = *(const v8h*)p;          // K 0..7   (lanes<16)  / 8..15  (lanes>=16)
  v8h hh = *(const v8h*)(p + 16);   // K 16..23 (lanes<16)  / 24..31 (lanes>=16)
  v16h r;
#pragma unroll
  for (int i = 0; i < 8; ++i) { r[i] = lo[i]; r[8 + i] = hh[i]; }
  return r;
}

__device__ __forceinline__ v8f wmma16(v16h a, v16h b, v8f c) {
  return __builtin_amdgcn_wmma_f32_16x16x32_f16(false, a, false, b,
                                                (short)0, c, false, false);
}

__device__ __forceinline__ void wait_async0() {
#if __has_builtin(__builtin_amdgcn_s_wait_asynccnt)
  __builtin_amdgcn_s_wait_asynccnt(0);
#else
  asm volatile("s_wait_asynccnt 0x0" ::: "memory");
#endif
}

// ---------------------------------------------------------------------------
// Weight transpose + f32->f16: dst[o*rows_in + i] = src[i*cols_out + o]
// ---------------------------------------------------------------------------
__global__ void transpose_cvt_kernel(const float* __restrict__ src,
                                     _Float16* __restrict__ dst,
                                     int rows_in, int cols_out) {
  int idx = blockIdx.x * blockDim.x + threadIdx.x;
  if (idx >= rows_in * cols_out) return;
  int i = idx / cols_out;
  int o = idx % cols_out;
  dst[o * rows_in + i] = (_Float16)src[i * cols_out + o];
}

// ---------------------------------------------------------------------------
// QKV projection. 512 blocks, 128 thr (4 waves). Each block: one 16-token tile.
// q,k stored f16 row-major per (b,h): [(bh)*N + n]*32 + d ; SCALE folded into q.
// v stored transposed per (b,h): vt[((bh)*32 + d)*N + n] (contiguous keys).
// ---------------------------------------------------------------------------
__global__ __launch_bounds__(128) void qkv_kernel(
    const float* __restrict__ x,
    const _Float16* __restrict__ wqT, const _Float16* __restrict__ wkT,
    const _Float16* __restrict__ wvT,
    const float* __restrict__ bq, const float* __restrict__ bk,
    const float* __restrict__ bv,
    _Float16* __restrict__ q, _Float16* __restrict__ k,
    _Float16* __restrict__ vt) {
  __shared__ __align__(16) _Float16 xa[16 * DIM];   // x tile, f16
  __shared__ __align__(16) _Float16 xp[16 * 96];    // channel-pooled tile
  int tok0 = blockIdx.x * 16;                        // flat token index b*N+n
  for (int e = threadIdx.x; e < 16 * DIM; e += 128) {
    int r = e / DIM, c = e % DIM;
    xa[e] = (_Float16)x[(size_t)(tok0 + r) * DIM + c];
  }
  for (int e = threadIdx.x; e < 16 * 96; e += 128) {
    int r = e / 96, c = e % 96;
    const float* p = &x[(size_t)(tok0 + r) * DIM + c * 4];
    xp[e] = (_Float16)(0.25f * (p[0] + p[1] + p[2] + p[3]));
  }
  __syncthreads();

  // readfirstlane -> SGPR: keeps job/kind/Kin scalar so the WMMA K-loop and
  // the per-kind stores are scalar branches (EXEC stays all-1s around WMMA).
  int wave = __builtin_amdgcn_readfirstlane(threadIdx.x >> 5);
  int lane = threadIdx.x & 31;
  int col = lane & 15, hi = lane >> 4;
  for (int job = wave; job < 72; job += 4) {   // 24 q + 24 k + 24 v col-tiles
    int kind = job / 24, jt = job % 24;
    const _Float16* wT  = (kind == 0) ? wqT : (kind == 1) ? wkT : wvT;
    const float*    bia = (kind == 0) ? bq  : (kind == 1) ? bk  : bv;
    const _Float16* xin = (kind == 0) ? xa  : xp;
    int Kin = (kind == 0) ? DIM : 96;
    v8f acc = {};
    for (int kc = 0; kc < Kin; kc += 32) {
      v16h a = load_tile16(xin + kc, Kin);
      v16h b = load_tile16(wT + (size_t)jt * 16 * Kin + kc, Kin);
      __builtin_prefetch(wT + (size_t)jt * 16 * Kin + kc + 32, 0, 1);
      acc = wmma16(a, b, acc);
    }
    int ocol = jt * 16 + col;
    float bb = bia[ocol];
    int h = ocol >> 5, d = ocol & 31;
#pragma unroll
    for (int i = 0; i < 8; ++i) {
      int gt = tok0 + i + 8 * hi;
      int b_ = gt >> 10, n = gt & 1023;
      float val = acc[i] + bb;
      if (kind == 0)
        q[((size_t)(b_ * NH + h) * NTOK + n) * HD + d] = (_Float16)(val * SCALE);
      else if (kind == 1)
        k[((size_t)(b_ * NH + h) * NTOK + n) * HD + d] = (_Float16)val;
      else
        vt[((size_t)(b_ * NH + h) * HD + d) * NTOK + n] = (_Float16)val;
    }
  }
}

// ---------------------------------------------------------------------------
// Flash attention. grid (B*H, N/64), 4 waves x 16 query rows. Online softmax;
// S/P never leave the wave except a 1KB LDS bounce to re-layout C->A.
// ---------------------------------------------------------------------------
__global__ __launch_bounds__(128) void attn_kernel(
    const _Float16* __restrict__ q, const _Float16* __restrict__ k,
    const _Float16* __restrict__ vt, _Float16* __restrict__ ctx) {
  __shared__ __align__(16) _Float16 pbuf[4 * 16 * 32];
  int bh = blockIdx.x, b = bh / NH, h = bh % NH;
  int wave = __builtin_amdgcn_readfirstlane(threadIdx.x >> 5);
  int lane = threadIdx.x & 31;
  int hi = lane >> 4, col = lane & 15;
  int qrow0 = blockIdx.y * 64 + wave * 16;
  const _Float16* qbh = q  + (size_t)bh * NTOK * HD;
  const _Float16* kbh = k  + (size_t)bh * NTOK * HD;
  const _Float16* vbh = vt + (size_t)bh * HD * NTOK;
  _Float16* pl = pbuf + wave * 16 * 32;

  v16h qa = load_tile16(qbh + (size_t)qrow0 * HD, HD);  // SCALE pre-folded
  v8f o0 = {}, o1 = {};
  float m[8], l[8];
#pragma unroll
  for (int i = 0; i < 8; ++i) { m[i] = -3.0e38f; l[i] = 0.f; }

  for (int kk = 0; kk < NTOK; kk += 32) {
    v16h kb0 = load_tile16(kbh + (size_t)kk * HD, HD);
    v16h kb1 = load_tile16(kbh + (size_t)(kk + 16) * HD, HD);
    v8f s0 = {}, s1 = {};
    s0 = wmma16(qa, kb0, s0);          // S[16q x keys kk..kk+15]
    s1 = wmma16(qa, kb1, s1);          // S[16q x keys kk+16..kk+31]
#pragma unroll
    for (int i = 0; i < 8; ++i) {      // row = i + 8*hi ; cols = lanes%16
      float r = fmaxf(s0[i], s1[i]);
      r = fmaxf(r, __shfl_xor(r, 1, 32));
      r = fmaxf(r, __shfl_xor(r, 2, 32));
      r = fmaxf(r, __shfl_xor(r, 4, 32));
      r = fmaxf(r, __shfl_xor(r, 8, 32));          // rowmax within 16-lane half
      float mn = fmaxf(m[i], r);
      float sc = __expf(m[i] - mn);
      m[i] = mn; l[i] *= sc; o0[i] *= sc; o1[i] *= sc;
      float p0 = __expf(s0[i] - mn), p1 = __expf(s1[i] - mn);
      float rs = p0 + p1;
      rs += __shfl_xor(rs, 1, 32); rs += __shfl_xor(rs, 2, 32);
      rs += __shfl_xor(rs, 4, 32); rs += __shfl_xor(rs, 8, 32);
      l[i] += rs;
      int row = i + 8 * hi;
      pl[row * 32 + col]      = (_Float16)p0;       // C-layout -> row-major LDS
      pl[row * 32 + 16 + col] = (_Float16)p1;
    }
    asm volatile("s_wait_dscnt 0" ::: "memory");    // CDNA5 split DS counter
    v16h pa  = load_tile16(pl, 32);                         // P as A (16x32)
    v16h vb0 = load_tile16(vbh + kk, NTOK);                 // V dims 0..15
    v16h vb1 = load_tile16(vbh + (size_t)16 * NTOK + kk, NTOK); // dims 16..31
    o0 = wmma16(pa, vb0, o0);
    o1 = wmma16(pa, vb1, o1);
  }
#pragma unroll
  for (int i = 0; i < 8; ++i) {
    float inv = 1.0f / l[i];
    int tokn = qrow0 + i + 8 * hi;
    size_t base = ((size_t)b * NTOK + tokn) * DIM + h * HD;
    ctx[base + col]      = (_Float16)(o0[i] * inv);
    ctx[base + 16 + col] = (_Float16)(o1[i] * inv);
  }
}

// ---------------------------------------------------------------------------
// Output projection: ctx(f16) @ WpT + bp -> f32 out.
// ctx tile staged via GLOBAL_LOAD_ASYNC_TO_LDS_B128 (ASYNCcnt) when available.
// ---------------------------------------------------------------------------
__global__ __launch_bounds__(128) void proj_kernel(
    const _Float16* __restrict__ ctx, const _Float16* __restrict__ wpT,
    const float* __restrict__ bp, float* __restrict__ out) {
  __shared__ __align__(16) _Float16 ca[16 * DIM];
  int tok0 = blockIdx.x * 16;
#if USE_ASYNC_LDS
  {
    // 16*384 f16 = 12KB = 768 x b128 chunks; pure byte copy -> async DMA path.
    gv4i* gsrc = (gv4i*)(ctx + (size_t)tok0 * DIM);
    lv4i* ldst = (lv4i*)ca;
    for (int it = threadIdx.x; it < 768; it += 128)
      __builtin_amdgcn_global_load_async_to_lds_b128(gsrc + it, ldst + it, 0, 0);
    wait_async0();
  }
#else
  for (int e = threadIdx.x; e < 16 * DIM; e += 128)
    ca[e] = ctx[(size_t)tok0 * DIM + e];
#endif
  __syncthreads();
  int wave = __builtin_amdgcn_readfirstlane(threadIdx.x >> 5);
  int lane = threadIdx.x & 31;
  int hi = lane >> 4, col = lane & 15;
  for (int jt = wave; jt < 24; jt += 4) {
    v8f acc = {};
    for (int kc = 0; kc < DIM; kc += 32) {
      v16h a = load_tile16(ca + kc, DIM);
      v16h b = load_tile16(wpT + (size_t)jt * 16 * DIM + kc, DIM);
      __builtin_prefetch(wpT + (size_t)jt * 16 * DIM + kc + 32, 0, 1);
      acc = wmma16(a, b, acc);
    }
    int ocol = jt * 16 + col;
    float bb = bp[ocol];
#pragma unroll
    for (int i = 0; i < 8; ++i)
      out[(size_t)(tok0 + i + 8 * hi) * DIM + ocol] = acc[i] + bb;
  }
}

extern "C" void kernel_launch(void* const* d_in, const int* in_sizes, int n_in,
                              void* d_out, int out_size, void* d_ws, size_t ws_size,
                              hipStream_t stream) {
  const float* x  = (const float*)d_in[0];
  const float* Wq = (const float*)d_in[1];
  const float* bq = (const float*)d_in[2];
  const float* Wk = (const float*)d_in[3];
  const float* bk = (const float*)d_in[4];
  const float* Wv = (const float*)d_in[5];
  const float* bv = (const float*)d_in[6];
  const float* Wp = (const float*)d_in[7];
  const float* bp = (const float*)d_in[8];
  float* out = (float*)d_out;

  char* ws = (char*)d_ws;
  size_t off = 0;
  auto take = [&](size_t bytes) -> char* {
    char* p = ws + off;
    off = (off + bytes + 255) & ~(size_t)255;
    return p;
  };
  _Float16* wqT = (_Float16*)take((size_t)DIM * DIM * 2);
  _Float16* wkT = (_Float16*)take((size_t)DIM * 96 * 2);
  _Float16* wvT = (_Float16*)take((size_t)DIM * 96 * 2);
  _Float16* wpT = (_Float16*)take((size_t)DIM * DIM * 2);
  _Float16* qd  = (_Float16*)take((size_t)NB * NH * NTOK * HD * 2);
  _Float16* kd  = (_Float16*)take((size_t)NB * NH * NTOK * HD * 2);
  _Float16* vtd = (_Float16*)take((size_t)NB * NH * NTOK * HD * 2);
  _Float16* ctx = (_Float16*)take((size_t)NB * NTOK * DIM * 2);

  transpose_cvt_kernel<<<(DIM * DIM + 255) / 256, 256, 0, stream>>>(Wq, wqT, DIM, DIM);
  transpose_cvt_kernel<<<(96 * DIM + 255) / 256, 256, 0, stream>>>(Wk, wkT, 96, DIM);
  transpose_cvt_kernel<<<(96 * DIM + 255) / 256, 256, 0, stream>>>(Wv, wvT, 96, DIM);
  transpose_cvt_kernel<<<(DIM * DIM + 255) / 256, 256, 0, stream>>>(Wp, wpT, DIM, DIM);

  qkv_kernel<<<NB * NTOK / 16, 128, 0, stream>>>(x, wqT, wkT, wvT, bq, bk, bv,
                                                 qd, kd, vtd);
  attn_kernel<<<dim3(NB * NH, NTOK / 64), 128, 0, stream>>>(qd, kd, vtd, ctx);
  proj_kernel<<<NB * NTOK / 16, 128, 0, stream>>>(ctx, wpT, bp, out);
}
